// Attention_13649406067174
// MI455X (gfx1250) — matrix operand order repaired and checked by reading, not compiled, not run on hardware
//
#include <hip/hip_runtime.h>
#include <hip/hip_bf16.h>

typedef _Float16 v16h __attribute__((ext_vector_type(16)));
typedef _Float16 v8h  __attribute__((ext_vector_type(8)));
typedef float    v8f  __attribute__((ext_vector_type(8)));
typedef int      v4i  __attribute__((ext_vector_type(4)));

#define WMMA(a, b, c) __builtin_amdgcn_wmma_f32_16x16x32_f16(false, (a), false, (b), (short)0, (c), false, false)

// ---------------------------------------------------------------------------
// Fragment loaders (wave32, 16x16x32 f16 WMMA)
// A-matrix 16x32 layout: lanes 0-15 row m=lane, K 0..15;
// lanes 16-31 row m=lane-16, K 16..31.
__device__ __forceinline__ v16h load_fragA(const _Float16* base, int stride, int lane) {
    const int r = lane & 15;
    const int s = (lane >> 4) ? 16 : 0;
    const _Float16* p = base + (size_t)r * stride;
    v8h lo = *(const v8h*)(p + s);
    v8h hi = *(const v8h*)(p + 8 + s);
    v16h f;
#pragma unroll
    for (int e = 0; e < 8; ++e) { f[e] = lo[e]; f[8 + e] = hi[e]; }
    return f;
}

// B-matrix 32x16 layout: lane n holds column n, K=0..15 contiguous; lane n+16
// holds K=16..31. `base` points at B^T rows, so loads are contiguous.
__device__ __forceinline__ v16h load_fragB(const _Float16* base, int stride, int lane) {
    const int n = lane & 15;
    const int s = (lane >> 4) ? 16 : 0;
    const _Float16* p = base + (size_t)n * stride + s;
    v8h lo = *(const v8h*)(p);
    v8h hi = *(const v8h*)(p + 8);
    v16h f;
#pragma unroll
    for (int e = 0; e < 8; ++e) { f[e] = lo[e]; f[8 + e] = hi[e]; }
    return f;
}

// CDNA5 async global->LDS copy (16B per lane), tracked by ASYNCcnt.
// Low 32 bits of a generic shared pointer are the LDS byte address (ISA 10.2).
__device__ __forceinline__ void async_b128(unsigned lds_addr, const void* gaddr) {
    asm volatile("global_load_async_to_lds_b128 %0, %1, off"
                 :: "v"(lds_addr), "v"(gaddr) : "memory");
}
__device__ __forceinline__ void wait_async0() {
    asm volatile("s_wait_asynccnt 0x0" ::: "memory");
}

// ---------------------------------------------------------------------------
// 1) elementwise f32 -> f16
__global__ void cvt_f32_to_f16(const float* __restrict__ src, _Float16* __restrict__ dst, int n) {
    for (int i = blockIdx.x * blockDim.x + threadIdx.x; i < n; i += gridDim.x * blockDim.x)
        dst[i] = (_Float16)src[i];
}

// 2) transpose-convert weight: w[K][N] (f32) -> wt[N][K] (f16)
__global__ void cvt_transpose(const float* __restrict__ w, _Float16* __restrict__ wt, int K, int N) {
    int total = K * N;
    for (int i = blockIdx.x * blockDim.x + threadIdx.x; i < total; i += gridDim.x * blockDim.x) {
        int c = i / K, k = i - c * K;
        wt[i] = (_Float16)w[k * N + c];
    }
}

// ---------------------------------------------------------------------------
// 3) position-bias MLP: 961 rows, per-thread 32-wide chains (pure VALU, tiny)
__device__ __forceinline__ void ln_relu32(float* v, const float* g, const float* b) {
    float m = 0.f;
#pragma unroll
    for (int j = 0; j < 32; ++j) m += v[j];
    m *= (1.f / 32.f);
    float var = 0.f;
#pragma unroll
    for (int j = 0; j < 32; ++j) { float d = v[j] - m; var += d * d; }
    var *= (1.f / 32.f);
    float inv = rsqrtf(var + 1e-5f);
#pragma unroll
    for (int j = 0; j < 32; ++j) {
        float y = (v[j] - m) * inv * g[j] + b[j];
        v[j] = y > 0.f ? y : 0.f;
    }
}

__global__ void pos_mlp(const float* __restrict__ biases,
                        const float* __restrict__ p1_w, const float* __restrict__ p1_b,
                        const float* __restrict__ g1, const float* __restrict__ b1,
                        const float* __restrict__ p2_w, const float* __restrict__ p2_b,
                        const float* __restrict__ g2, const float* __restrict__ b2,
                        const float* __restrict__ p3_w, const float* __restrict__ p3_b,
                        const float* __restrict__ g3, const float* __restrict__ b3,
                        const float* __restrict__ p4_w, const float* __restrict__ p4_b,
                        float* __restrict__ pos) {
    int row = blockIdx.x * blockDim.x + threadIdx.x;
    if (row >= 961) return;
    float x0 = biases[row * 2 + 0], x1 = biases[row * 2 + 1];
    float a[32], t[32];
#pragma unroll
    for (int j = 0; j < 32; ++j) a[j] = x0 * p1_w[j] + x1 * p1_w[32 + j] + p1_b[j];
    ln_relu32(a, g1, b1);
    for (int o = 0; o < 32; ++o) {
        float acc = p2_b[o];
#pragma unroll
        for (int j = 0; j < 32; ++j) acc += a[j] * p2_w[j * 32 + o];
        t[o] = acc;
    }
    ln_relu32(t, g2, b2);
    for (int o = 0; o < 32; ++o) {
        float acc = p3_b[o];
#pragma unroll
        for (int j = 0; j < 32; ++j) acc += t[j] * p3_w[j * 32 + o];
        a[o] = acc;
    }
    ln_relu32(a, g3, b3);
    for (int o = 0; o < 8; ++o) {
        float acc = p4_b[o];
#pragma unroll
        for (int j = 0; j < 32; ++j) acc += a[j] * p4_w[j * 8 + o];
        pos[row * 8 + o] = acc;
    }
}

// ---------------------------------------------------------------------------
// 4) qkv GEMM (pipelined + fully unrolled K, 32Mx64N per wave):
//    xh[65536,512] @ qkv_w + qkv_b ; q scaled by hd^-0.5
//    layouts: q,k -> [B,H,256,64] f16 ; v -> vT [B,H,64,256] f16
__global__ __launch_bounds__(256) void qkv_gemm(
        const _Float16* __restrict__ xh, const _Float16* __restrict__ wt,
        const float* __restrict__ qkv_b,
        _Float16* __restrict__ q, _Float16* __restrict__ k, _Float16* __restrict__ vT) {
    const int lane = threadIdx.x & 31;
    const int wave = threadIdx.x >> 5;
    const int m0 = blockIdx.x * 128 + (wave & 3) * 32;
    const int n0 = blockIdx.y * 128 + (wave >> 2) * 64;
    const _Float16* aBase = xh + (size_t)m0 * 512;
    const _Float16* bBase = wt + (size_t)n0 * 512;

    v8f acc[2][4] = {};
    v16h a0 = load_fragA(aBase, 512, lane);
    v16h a1 = load_fragA(aBase + (size_t)16 * 512, 512, lane);
    v16h b[4];
#pragma unroll
    for (int t = 0; t < 4; ++t) b[t] = load_fragB(bBase + (size_t)t * 16 * 512, 512, lane);

#pragma unroll
    for (int kc = 0; kc < 480; kc += 32) {
        const int kn = kc + 32;
        v16h a0n = load_fragA(aBase + kn, 512, lane);
        v16h a1n = load_fragA(aBase + (size_t)16 * 512 + kn, 512, lane);
        v16h bn[4];
#pragma unroll
        for (int t = 0; t < 4; ++t) bn[t] = load_fragB(bBase + (size_t)t * 16 * 512 + kn, 512, lane);
#pragma unroll
        for (int t = 0; t < 4; ++t) {
            acc[0][t] = WMMA(a0, b[t], acc[0][t]);
            acc[1][t] = WMMA(a1, b[t], acc[1][t]);
        }
        a0 = a0n; a1 = a1n;
#pragma unroll
        for (int t = 0; t < 4; ++t) b[t] = bn[t];
    }
#pragma unroll
    for (int t = 0; t < 4; ++t) {
        acc[0][t] = WMMA(a0, b[t], acc[0][t]);
        acc[1][t] = WMMA(a1, b[t], acc[1][t]);
    }

    const int r0 = (lane >> 4) ? 8 : 0;
#pragma unroll
    for (int s = 0; s < 2; ++s) {
        const int ms = m0 + s * 16;
#pragma unroll
        for (int t = 0; t < 4; ++t) {
            const int c = n0 + t * 16 + (lane & 15);
            const int which = c >> 9;        // 0=q 1=k 2=v
            const int h = (c & 511) >> 6;
            const int d = c & 63;
            const float bias = qkv_b[c];
            if (which == 2) {
                v8h pk;
#pragma unroll
                for (int r = 0; r < 8; ++r) pk[r] = (_Float16)(acc[s][t][r] + bias);
                const int tok0 = ms + r0;
                const int b_ = tok0 >> 8, n_ = tok0 & 255;
                *(v8h*)(vT + ((size_t)(b_ * 8 + h) * 64 + d) * 256 + n_) = pk;
            } else {
                const float sc = (which == 0) ? 0.125f : 1.0f;
                _Float16* dst = (which == 0) ? q : k;
#pragma unroll
                for (int r = 0; r < 8; ++r) {
                    const int tok = ms + r0 + r;
                    const int b_ = tok >> 8, n_ = tok & 255;
                    dst[((size_t)(b_ * 8 + h) * 256 + n_) * 64 + d] = (_Float16)((acc[s][t][r] + bias) * sc);
                }
            }
        }
    }
}

// ---------------------------------------------------------------------------
// 5) attention per (b,h): S^T = k @ q^T so softmax rows are lane-local and the
//    exp'd accumulators become O^T B-fragments in-register (one shfl_xor(16)).
__global__ __launch_bounds__(512) void attention(
        const _Float16* __restrict__ q, const _Float16* __restrict__ k,
        const _Float16* __restrict__ vT, const float* __restrict__ pos,
        const int* __restrict__ rel_idx, _Float16* __restrict__ o) {
    extern __shared__ __align__(16) char smem_raw[];
    _Float16* kS = (_Float16*)smem_raw;        // [256][64]
    _Float16* vS = kS + 256 * 64;              // [64][256]

    const int bh = blockIdx.x;                 // b*8 + h
    const int h = bh & 7;
    const int lane = threadIdx.x & 31;
    const int wave = threadIdx.x >> 5;
    const bool hiHalf = (lane >= 16);
    const int r0 = hiHalf ? 8 : 0;

    {   // stage k and vT tiles (32KB each) into LDS via CDNA5 async copies
        const unsigned kL = (unsigned)(size_t)kS;
        const unsigned vL = (unsigned)(size_t)vS;
        const _Float16* kG = k + (size_t)bh * 256 * 64;
        const _Float16* vG = vT + (size_t)bh * 64 * 256;
#pragma unroll
        for (int i = 0; i < 4; ++i) {
            const int e = threadIdx.x + i * 512;
            async_b128(kL + e * 16, kG + (size_t)e * 8);
            async_b128(vL + e * 16, vG + (size_t)e * 8);
        }
        wait_async0();
    }
    __syncthreads();

    const int i0 = wave * 16;                  // this wave's 16 query rows
    const _Float16* qrow = q + ((size_t)bh * 256 + i0) * 64;
    v16h qf0 = load_fragB(qrow + 0, 64, lane);
    v16h qf1 = load_fragB(qrow + 32, 64, lane);

    // S^T tiles: 16 key tiles x (this wave's query tile)
    v8f s[16];
#pragma unroll
    for (int jt = 0; jt < 16; ++jt) {
        v16h a0 = load_fragA(kS + jt * 16 * 64 + 0, 64, lane);
        v16h a1 = load_fragA(kS + jt * 16 * 64 + 32, 64, lane);
        v8f c = {};
        c = WMMA(a0, qf0, c);
        c = WMMA(a1, qf1, c);
        s[jt] = c;
    }

    // add relative position bias: element (j = jt*16+r0+r, i = i0+lane%16)
    const int iq = i0 + (lane & 15);
#pragma unroll
    for (int jt = 0; jt < 16; ++jt) {
        const v4i* rp = (const v4i*)(rel_idx + (size_t)iq * 256 + jt * 16 + r0);
        v4i ia = rp[0], ib = rp[1];
#pragma unroll
        for (int r = 0; r < 8; ++r) {
            int idx = (r < 4) ? ia[r] : ib[r - 4];
            s[jt][r] += pos[idx * 8 + h];
        }
    }

    // softmax along j (row of S = column of S^T, split across lane pair l / l+16)
    float mx = -3.402823466e+38f;
#pragma unroll
    for (int jt = 0; jt < 16; ++jt)
#pragma unroll
        for (int r = 0; r < 8; ++r) mx = fmaxf(mx, s[jt][r]);
    mx = fmaxf(mx, __shfl_xor(mx, 16));
    float sum = 0.f;
#pragma unroll
    for (int jt = 0; jt < 16; ++jt)
#pragma unroll
        for (int r = 0; r < 8; ++r) { float e = __expf(s[jt][r] - mx); s[jt][r] = e; sum += e; }
    sum += __shfl_xor(sum, 16);
    const float inv = 1.f / sum;

    // O^T = vT @ P^T : 4 d-tiles, 8 token chunks of 32
    v8f oacc[4] = {};
#pragma unroll
    for (int c = 0; c < 8; ++c) {
        v8f ea = s[2 * c], eb = s[2 * c + 1];
        v16h pf;
#pragma unroll
        for (int e = 0; e < 8; ++e) {
            float oa = __shfl_xor(ea[e], 16);
            float ob = __shfl_xor(eb[e], 16);
            pf[e]     = hiHalf ? (_Float16)ob    : (_Float16)ea[e];
            pf[8 + e] = hiHalf ? (_Float16)eb[e] : (_Float16)oa;
        }
#pragma unroll
        for (int dt = 0; dt < 4; ++dt) {
            v16h a = load_fragA(vS + dt * 16 * 256 + c * 32, 256, lane);
            oacc[dt] = WMMA(a, pf, oacc[dt]);
        }
    }

    // normalize + store: o layout [B,H,256,64] f16
#pragma unroll
    for (int dt = 0; dt < 4; ++dt) {
        v8h pk;
#pragma unroll
        for (int r = 0; r < 8; ++r) pk[r] = (_Float16)(oacc[dt][r] * inv);
        *(v8h*)(o + ((size_t)bh * 256 + iq) * 64 + dt * 16 + r0) = pk;
    }
}

// ---------------------------------------------------------------------------
// 6) proj GEMM (pipelined + fully unrolled K, 32Mx64N per wave):
//    ao([B,H,256,64] f16, logical [65536,512]) @ proj_w + proj_b -> f32 out
__device__ __forceinline__ v16h load_a_ao(const _Float16* ao, int tok, int kc, int lane) {
    const int hiSel = (lane >> 4) ? 16 : 0;
    const size_t b_ = (unsigned)tok >> 8;
    const int n_ = tok & 255;
    const int c0 = kc + hiSel, c1 = c0 + 8;
    v8h lo = *(const v8h*)(ao + ((b_ * 8 + (c0 >> 6)) * 256 + n_) * 64 + (c0 & 63));
    v8h hi = *(const v8h*)(ao + ((b_ * 8 + (c1 >> 6)) * 256 + n_) * 64 + (c1 & 63));
    v16h a;
#pragma unroll
    for (int e = 0; e < 8; ++e) { a[e] = lo[e]; a[8 + e] = hi[e]; }
    return a;
}

__global__ __launch_bounds__(256) void proj_gemm(
        const _Float16* __restrict__ ao, const _Float16* __restrict__ wt,
        const float* __restrict__ proj_b, float* __restrict__ out) {
    const int lane = threadIdx.x & 31;
    const int wave = threadIdx.x >> 5;
    const int m0 = blockIdx.x * 128 + (wave & 3) * 32;
    const int n0 = blockIdx.y * 128 + (wave >> 2) * 64;
    const int tok0 = m0 + (lane & 15);
    const _Float16* bBase = wt + (size_t)n0 * 512;

    v8f acc[2][4] = {};
    v16h a0 = load_a_ao(ao, tok0, 0, lane);
    v16h a1 = load_a_ao(ao, tok0 + 16, 0, lane);
    v16h b[4];
#pragma unroll
    for (int t = 0; t < 4; ++t) b[t] = load_fragB(bBase + (size_t)t * 16 * 512, 512, lane);

#pragma unroll
    for (int kc = 0; kc < 480; kc += 32) {
        const int kn = kc + 32;
        v16h a0n = load_a_ao(ao, tok0, kn, lane);
        v16h a1n = load_a_ao(ao, tok0 + 16, kn, lane);
        v16h bn[4];
#pragma unroll
        for (int t = 0; t < 4; ++t) bn[t] = load_fragB(bBase + (size_t)t * 16 * 512 + kn, 512, lane);
#pragma unroll
        for (int t = 0; t < 4; ++t) {
            acc[0][t] = WMMA(a0, b[t], acc[0][t]);
            acc[1][t] = WMMA(a1, b[t], acc[1][t]);
        }
        a0 = a0n; a1 = a1n;
#pragma unroll
        for (int t = 0; t < 4; ++t) b[t] = bn[t];
    }
#pragma unroll
    for (int t = 0; t < 4; ++t) {
        acc[0][t] = WMMA(a0, b[t], acc[0][t]);
        acc[1][t] = WMMA(a1, b[t], acc[1][t]);
    }

    const int r0 = (lane >> 4) ? 8 : 0;
#pragma unroll
    for (int s = 0; s < 2; ++s) {
#pragma unroll
        for (int t = 0; t < 4; ++t) {
            const int c = n0 + t * 16 + (lane & 15);
            const float bias = proj_b[c];
#pragma unroll
            for (int r = 0; r < 8; ++r) {
                const int tk = m0 + s * 16 + r0 + r;
                out[(size_t)tk * 512 + c] = acc[s][t][r] + bias;
            }
        }
    }
}

// ---------------------------------------------------------------------------
extern "C" void kernel_launch(void* const* d_in, const int* in_sizes, int n_in,
                              void* d_out, int out_size, void* d_ws, size_t ws_size,
                              hipStream_t stream) {
    const float* x      = (const float*)d_in[0];
    const float* qkv_w  = (const float*)d_in[1];
    const float* qkv_b  = (const float*)d_in[2];
    const float* proj_w = (const float*)d_in[3];
    const float* proj_b = (const float*)d_in[4];
    const float* p1_w = (const float*)d_in[5];  const float* p1_b = (const float*)d_in[6];
    const float* g1   = (const float*)d_in[7];  const float* b1   = (const float*)d_in[8];
    const float* p2_w = (const float*)d_in[9];  const float* p2_b = (const float*)d_in[10];
    const float* g2   = (const float*)d_in[11]; const float* b2   = (const float*)d_in[12];
    const float* p3_w = (const float*)d_in[13]; const float* p3_b = (const float*)d_in[14];
    const float* g3   = (const float*)d_in[15]; const float* b3   = (const float*)d_in[16];
    const float* p4_w = (const float*)d_in[17]; const float* p4_b = (const float*)d_in[18];
    const float* biases = (const float*)d_in[19];
    const int* rel_idx  = (const int*)d_in[20];
    float* out = (float*)d_out;

    // workspace carve-up (all offsets 256B-aligned by construction)
    char* w = (char*)d_ws;
    _Float16* xh     = (_Float16*)w; w += (size_t)65536 * 512 * 2;   // 64 MiB
    _Float16* wqkvT  = (_Float16*)w; w += (size_t)1536 * 512 * 2;    // 1.5 MiB
    _Float16* wprojT = (_Float16*)w; w += (size_t)512 * 512 * 2;     // 0.5 MiB
    _Float16* qws    = (_Float16*)w; w += (size_t)2048 * 256 * 64 * 2;
    _Float16* kws    = (_Float16*)w; w += (size_t)2048 * 256 * 64 * 2;
    _Float16* vTws   = (_Float16*)w; w += (size_t)2048 * 64 * 256 * 2;
    _Float16* ows    = (_Float16*)w; w += (size_t)2048 * 256 * 64 * 2;
    float* pos       = (float*)w;    w += (size_t)961 * 8 * 4;

    cvt_f32_to_f16<<<4096, 256, 0, stream>>>(x, xh, 65536 * 512);
    cvt_transpose<<<1536, 256, 0, stream>>>(qkv_w, wqkvT, 512, 1536);
    cvt_transpose<<<512, 256, 0, stream>>>(proj_w, wprojT, 512, 512);
    pos_mlp<<<4, 256, 0, stream>>>(biases, p1_w, p1_b, g1, b1, p2_w, p2_b, g2, b2,
                                   p3_w, p3_b, g3, b3, p4_w, p4_b, pos);
    qkv_gemm<<<dim3(512, 12), 256, 0, stream>>>(xh, wqkvT, qkv_b, qws, kws, vTws);
    attention<<<2048, 512, 65536, stream>>>(qws, kws, vTws, pos, rel_idx, ows);
    proj_gemm<<<dim3(512, 4), 256, 0, stream>>>(ows, wprojT, proj_b, out);
}
